// EnhancedHybridPerceptionBlock_49692771615256
// MI455X (gfx1250) — compile-verified, matrix operand order checked by reading, not run on hardware
//
#include <hip/hip_runtime.h>
#include <cstdint>

// ---------------------------------------------------------------------------
// EnhancedHybridPerceptionBlock for MI455X (gfx1250, wave32, WMMA bf16)
// ---------------------------------------------------------------------------
#define BATCH  2
#define CH     256
#define IMGH   48
#define IMGW   48
#define NPIX   (IMGH*IMGW)        // 2304
#define BNROWS (BATCH*NPIX)       // 4608
#define NHEADS 8
#define DHEAD  32
#define NBH    (BATCH*NHEADS)     // 16
#define TOPK   576
#define HID    1024
#define GCH    32

typedef __bf16 bf16;
typedef __attribute__((ext_vector_type(16))) __bf16 v16bf;
typedef __attribute__((ext_vector_type(8)))  float  v8f;

__device__ __forceinline__ v8f v8f_zero() {
  v8f z;
#pragma unroll
  for (int i = 0; i < 8; ++i) z[i] = 0.0f;
  return z;
}

// Load a 16x32 bf16 fragment (A-layout == B-layout for row-major tiles):
// lane l holds row (l&15); elems 0..7 -> K = kb+0..7, elems 8..15 -> K = kb+16..23,
// kb = (l>=16) ? 8 : 0.   (CDNA5 ISA 7.12.2, 16-bit A-matrix 16x32)
__device__ __forceinline__ v16bf frag_ld(const bf16* base, int strideElems) {
  int lane = threadIdx.x & 31;
  const bf16* p = base + (lane & 15) * strideElems + ((lane >> 4) << 3);
  v16bf f;
#pragma unroll
  for (int i = 0; i < 8; ++i) f[i] = p[i];
#pragma unroll
  for (int i = 0; i < 8; ++i) f[i + 8] = p[i + 16];
  return f;
}

__device__ __forceinline__ v8f wmma_bf16(v16bf a, v16bf b, v8f c) {
  return __builtin_amdgcn_wmma_f32_16x16x32_bf16(false, a, false, b, (short)0, c,
                                                 false, false);
}

__device__ __forceinline__ float gelu_exact(float x) {
  return 0.5f * x * (1.0f + erff(x * 0.70710678118654752f));
}
__device__ __forceinline__ float sigmoidf(float x) { return 1.0f / (1.0f + __expf(-x)); }
__device__ __forceinline__ float siluf(float x)    { return x * sigmoidf(x); }

// ---------------------------------------------------------------------------
// f32 -> bf16 convert (plain, and transposing [K,N] -> [N,K])
// ---------------------------------------------------------------------------
__global__ void ehpb_f2b(const float* __restrict__ s, bf16* __restrict__ d, int n) {
  int i = blockIdx.x * blockDim.x + threadIdx.x;
  if (i < n) d[i] = (bf16)s[i];
}

__global__ void ehpb_f2bt(const float* __restrict__ s, bf16* __restrict__ d,
                          int K, int N) {
  int i = blockIdx.x * blockDim.x + threadIdx.x;
  if (i < K * N) {
    int k = i / N, n = i % N;
    d[(long)n * K + k] = (bf16)s[i];
  }
}

// ---------------------------------------------------------------------------
// LayerNorm over channels of NCHW tensor -> row-major [BN, C] bf16
// ---------------------------------------------------------------------------
__global__ __launch_bounds__(256)
void ehpb_ln_nchw(const float* __restrict__ src, const float* __restrict__ g,
                  const float* __restrict__ b, bf16* __restrict__ out) {
  int bn = blockIdx.x, t = threadIdx.x;
  int bb = bn / NPIX, n = bn % NPIX;
  float v = src[(bb * CH + t) * NPIX + n];
  __shared__ float red[256];
  red[t] = v; __syncthreads();
  for (int s = 128; s > 0; s >>= 1) { if (t < s) red[t] += red[t + s]; __syncthreads(); }
  float mu = red[0] * (1.0f / CH); __syncthreads();
  float d = v - mu;
  red[t] = d * d; __syncthreads();
  for (int s = 128; s > 0; s >>= 1) { if (t < s) red[t] += red[t + s]; __syncthreads(); }
  float var = red[0] * (1.0f / CH);
  float y = d * rsqrtf(var + 1e-6f) * g[t] + b[t];
  out[bn * CH + t] = (bf16)y;
}

// ---------------------------------------------------------------------------
// Generic WMMA GEMM:  C[M,N] = act(A[M,K](bf16) x BT[N,K](bf16)^T + bias[N])
// B operand is PRE-TRANSPOSED ([N][K], ld = K) so both tiles stage via
// global_load_async_to_lds_b128.  Block tile 64(M) x 128(N); 8 waves as 2x4,
// each wave a 32x32 accumulator (4 WMMAs / K-step); double-buffered LDS.
// Inner loop: issue 3 async DMA / thread -> 4 frag loads -> 4 WMMA ->
// s_wait_asynccnt 0 -> barrier.
// ---------------------------------------------------------------------------
__global__ __launch_bounds__(256)
void ehpb_gemm(const bf16* __restrict__ A, const bf16* __restrict__ Bm,
               const float* __restrict__ bias, float* __restrict__ Cf,
               bf16* __restrict__ Cb, int M, int Nn, int K, int act,
               long aStride, long bStride, long cStride) {
  A  += (long)blockIdx.z * aStride;
  Bm += (long)blockIdx.z * bStride;
  __shared__ __align__(16) bf16 As[2][64 * 40];   // [row][k], stride 40
  __shared__ __align__(16) bf16 Bs[2][128 * 40];  // [n][k],   stride 40
  int t = threadIdx.x;
  int w = t >> 5, lane = t & 31;
  int wm = w >> 2, wn = w & 3;
  int m0 = blockIdx.y * 64, n0 = blockIdx.x * 128;

  // stage one 64x32 A tile + 128x32 B^T tile, all async DMA -> LDS
  auto stage = [&](int k0, int buf) {
    int rowA = t >> 2, segA = (t & 3) * 8;              // 256 lanes x 16B == 4KB
    const bf16* ga = A + (long)(m0 + rowA) * K + k0 + segA;
    unsigned ldsA = (unsigned)(uintptr_t)&As[buf][rowA * 40 + segA];
    asm volatile("global_load_async_to_lds_b128 %0, %1, off"
                 :: "v"(ldsA), "v"(ga) : "memory");
#pragma unroll
    for (int j = 0; j < 2; ++j) {                       // 512 x 16B == 8KB
      int e = t + 256 * j;
      int rowB = e >> 2, segB = (e & 3) * 8;
      const bf16* gb = Bm + (long)(n0 + rowB) * K + k0 + segB;
      unsigned ldsB = (unsigned)(uintptr_t)&Bs[buf][rowB * 40 + segB];
      asm volatile("global_load_async_to_lds_b128 %0, %1, off"
                   :: "v"(ldsB), "v"(gb) : "memory");
    }
  };

  v8f acc[2][2];
  acc[0][0] = v8f_zero(); acc[0][1] = v8f_zero();
  acc[1][0] = v8f_zero(); acc[1][1] = v8f_zero();

  int nsteps = K >> 5;
  stage(0, 0);
  asm volatile("s_wait_asynccnt 0" ::: "memory");
  __syncthreads();

  for (int s = 0; s < nsteps; ++s) {
    int buf = s & 1;
    if (s + 1 < nsteps) stage((s + 1) * 32, buf ^ 1);
    if (s + 2 < nsteps)
      __builtin_prefetch(Bm + (long)(n0 + (t >> 1)) * K + (s + 2) * 32, 0, 1);
    v16bf fa0 = frag_ld(&As[buf][(wm * 32) * 40], 40);
    v16bf fa1 = frag_ld(&As[buf][(wm * 32 + 16) * 40], 40);
    v16bf fb0 = frag_ld(&Bs[buf][(wn * 32) * 40], 40);
    v16bf fb1 = frag_ld(&Bs[buf][(wn * 32 + 16) * 40], 40);
    acc[0][0] = wmma_bf16(fa0, fb0, acc[0][0]);
    acc[0][1] = wmma_bf16(fa0, fb1, acc[0][1]);
    acc[1][0] = wmma_bf16(fa1, fb0, acc[1][0]);
    acc[1][1] = wmma_bf16(fa1, fb1, acc[1][1]);
    asm volatile("s_wait_asynccnt 0" ::: "memory");
    __syncthreads();
  }

  // epilogue: C layout lane<16 -> M=r, lane>=16 -> M=8+r; N = lane&15
  int col = lane & 15, rb = (lane >> 4) << 3;
  long cOff = (long)blockIdx.z * cStride;
#pragma unroll
  for (int j = 0; j < 2; ++j) {
    int gn = n0 + wn * 32 + j * 16 + col;
    float bv = bias ? bias[gn] : 0.0f;
#pragma unroll
    for (int i = 0; i < 2; ++i) {
      int gmb = m0 + wm * 32 + i * 16 + rb;
#pragma unroll
      for (int r = 0; r < 8; ++r) {
        float v = acc[i][j][r] + bv;
        if (act == 1) v = gelu_exact(v);
        long idx = cOff + (long)(gmb + r) * Nn + gn;
        if (Cf) Cf[idx] = v;
        if (Cb) Cb[idx] = (bf16)v;
      }
    }
  }
}

// ---------------------------------------------------------------------------
// Split qkv f32 [B,N,3C] into bf16 q/k [B,H,N,32] and V transposed [B,H,32,N]
// (dim-major V so attention's P.V B-fragments load contiguously from global)
// ---------------------------------------------------------------------------
__global__ void ehpb_repack(const float* __restrict__ qkv, bf16* __restrict__ qb,
                            bf16* __restrict__ kb, bf16* __restrict__ vbT) {
  int idx = blockIdx.x * blockDim.x + threadIdx.x;   // BN*CH threads
  int bn = idx >> 8, c = idx & 255;
  int b = bn / NPIX, n = bn % NPIX;
  int h = c >> 5, d = c & 31;
  long src = (long)bn * (3 * CH);
  int bh = b * NHEADS + h;
  long dst = ((long)bh * NPIX + n) * DHEAD + d;
  qb[dst] = (bf16)qkv[src + c];
  kb[dst] = (bf16)qkv[src + CH + c];
  vbT[((long)bh * DHEAD + d) * NPIX + n] = (bf16)qkv[src + 2 * CH + c];
}

// ---------------------------------------------------------------------------
// Top-k row/col masks.  row_score_i = q_i . mean(k);  col_score_j = k_j . mean(q).
// Threshold via binary search on value so that count(score>=thr) >= TOPK.
// One block per (b,h).
// ---------------------------------------------------------------------------
__global__ __launch_bounds__(256)
void ehpb_masks(const bf16* __restrict__ qb, const bf16* __restrict__ kb,
                float* __restrict__ rmask, float* __restrict__ cmask) {
  int bh = blockIdx.x, t = threadIdx.x;
  __shared__ float meanv[DHEAD];
  __shared__ float red[256];
  __shared__ int cnt;
  for (int pass = 0; pass < 2; ++pass) {
    const bf16* S = pass == 0 ? qb : kb;   // rows being scored
    const bf16* Mm = pass == 0 ? kb : qb;  // mean source
    float* mask = pass == 0 ? rmask : cmask;
    float loc[DHEAD];
#pragma unroll
    for (int d = 0; d < DHEAD; ++d) loc[d] = 0.0f;
    for (int i = t; i < NPIX; i += 256) {
      const bf16* row = Mm + ((long)bh * NPIX + i) * DHEAD;
#pragma unroll
      for (int d = 0; d < DHEAD; ++d) loc[d] += (float)row[d];
    }
    if (t < DHEAD) meanv[t] = 0.0f;
    __syncthreads();
#pragma unroll
    for (int d = 0; d < DHEAD; ++d) atomicAdd(&meanv[d], loc[d]);
    __syncthreads();
    float sc[NPIX / 256];
    int ns = 0;
    for (int i = t; i < NPIX; i += 256, ++ns) {
      const bf16* row = S + ((long)bh * NPIX + i) * DHEAD;
      float s = 0.0f;
#pragma unroll
      for (int d = 0; d < DHEAD; ++d) s += (float)row[d] * meanv[d];
      sc[ns] = s;
    }
    float mn = 1e30f, mx = -1e30f;
    for (int k = 0; k < ns; ++k) { mn = fminf(mn, sc[k]); mx = fmaxf(mx, sc[k]); }
    red[t] = mn; __syncthreads();
    for (int s = 128; s > 0; s >>= 1) { if (t < s) red[t] = fminf(red[t], red[t + s]); __syncthreads(); }
    float lo = red[0]; __syncthreads();
    red[t] = mx; __syncthreads();
    for (int s = 128; s > 0; s >>= 1) { if (t < s) red[t] = fmaxf(red[t], red[t + s]); __syncthreads(); }
    float hi = red[0] + 1.0f; __syncthreads();
    for (int it = 0; it < 24; ++it) {
      float mid = 0.5f * (lo + hi);
      if (t == 0) cnt = 0;
      __syncthreads();
      int c = 0;
      for (int k = 0; k < ns; ++k) c += (sc[k] >= mid) ? 1 : 0;
      atomicAdd(&cnt, c);
      __syncthreads();
      int tot = cnt;
      __syncthreads();
      if (tot >= TOPK) lo = mid; else hi = mid;
    }
    for (int i = t, k = 0; i < NPIX; i += 256, ++k)
      mask[bh * NPIX + i] = (sc[k] >= lo) ? 1.0f : 0.0f;
    __syncthreads();
  }
}

// ---------------------------------------------------------------------------
// Fused pruned attention (flash-style streaming softmax; masked entries keep
// score exactly 0, matching attn*mask then softmax).  One wave per 16 queries.
// QK^T and P.V via v_wmma_f32_16x16x32_bf16 (d_head==32==K).  V is consumed
// pre-transposed straight from global; only P round-trips LDS (C->A layout).
// ---------------------------------------------------------------------------
__global__ __launch_bounds__(256)
void ehpb_attn(const bf16* __restrict__ qb, const bf16* __restrict__ kb,
               const bf16* __restrict__ vbT, const float* __restrict__ rmask,
               const float* __restrict__ cmask, bf16* __restrict__ ob) {
  __shared__ __align__(16) bf16 Pst[8][16 * 40];  // per-wave P tile [16 q][32 k]
  int t = threadIdx.x, w = t >> 5, lane = t & 31;
  int gw = blockIdx.x * 8 + w;
  int bh = gw / (NPIX / 16), qt = gw % (NPIX / 16);
  int q0 = qt * 16;
  const float scale = 0.17677669529663689f;  // 32^-0.5

  v16bf fq = frag_ld(qb + ((long)bh * NPIX + q0) * DHEAD, DHEAD);
  const bf16* kbase = kb + (long)bh * NPIX * DHEAD;
  const bf16* vtb = vbT + (long)bh * DHEAD * NPIX;
  int rb = ((lane >> 4) << 3);
  float rowm[8], mrow[8], lsum[8];
#pragma unroll
  for (int r = 0; r < 8; ++r) {
    rowm[r] = rmask[bh * NPIX + q0 + rb + r];
    mrow[r] = -1e30f;
    lsum[r] = 0.0f;
  }
  v8f o0 = v8f_zero(), o1 = v8f_zero();

  for (int j0 = 0; j0 < NPIX; j0 += 32) {
    v16bf fk0 = frag_ld(kbase + (long)j0 * DHEAD, DHEAD);
    v16bf fk1 = frag_ld(kbase + (long)(j0 + 16) * DHEAD, DHEAD);
    if (j0 + 32 < NPIX)
      __builtin_prefetch(kbase + (long)(j0 + 32 + lane) * DHEAD, 0, 1);
    v8f s0 = wmma_bf16(fq, fk0, v8f_zero());
    v8f s1 = wmma_bf16(fq, fk1, v8f_zero());
    float cm0 = cmask[bh * NPIX + j0 + (lane & 15)];
    float cm1 = cmask[bh * NPIX + j0 + 16 + (lane & 15)];
    int cc = lane & 15;
#pragma unroll
    for (int r = 0; r < 8; ++r) {
      float a = s0[r] * scale * rowm[r] * cm0;   // pruned -> exactly 0
      float b = s1[r] * scale * rowm[r] * cm1;
      float mxv = fmaxf(a, b);
      mxv = fmaxf(mxv, __shfl_xor(mxv, 1));
      mxv = fmaxf(mxv, __shfl_xor(mxv, 2));
      mxv = fmaxf(mxv, __shfl_xor(mxv, 4));
      mxv = fmaxf(mxv, __shfl_xor(mxv, 8));
      float mnew = fmaxf(mrow[r], mxv);
      float corr = __expf(mrow[r] - mnew);
      mrow[r] = mnew;
      float p0 = __expf(a - mnew), p1 = __expf(b - mnew);
      float rs = p0 + p1;
      rs += __shfl_xor(rs, 1);
      rs += __shfl_xor(rs, 2);
      rs += __shfl_xor(rs, 4);
      rs += __shfl_xor(rs, 8);
      lsum[r] = lsum[r] * corr + rs;
      o0[r] *= corr;
      o1[r] *= corr;
      Pst[w][(rb + r) * 40 + cc] = (bf16)p0;
      Pst[w][(rb + r) * 40 + 16 + cc] = (bf16)p1;
    }
    asm volatile("s_wait_dscnt 0" ::: "memory");
    v16bf fp  = frag_ld(&Pst[w][0], 40);
    v16bf fv0 = frag_ld(vtb + j0, NPIX);              // dims 0..15
    v16bf fv1 = frag_ld(vtb + 16 * NPIX + j0, NPIX);  // dims 16..31
    o0 = wmma_bf16(fp, fv0, o0);
    o1 = wmma_bf16(fp, fv1, o1);
  }

  int b = bh >> 3, h = bh & 7;
#pragma unroll
  for (int r = 0; r < 8; ++r) {
    int n = q0 + rb + r;
    long base = ((long)(b * NPIX + n)) * CH + h * DHEAD + (lane & 15);
    ob[base] = (bf16)(o0[r] / lsum[r]);
    ob[base + 16] = (bf16)(o1[r] / lsum[r]);
  }
}

// ---------------------------------------------------------------------------
// Depthwise 3x3 conv + bias, NCHW f32 -> bf16 written TRANSPOSED per batch
// as [n][ci] (= B^T operand for the pointwise-conv GEMM).
// Thread mapping c-fastest so the 2B stores coalesce.
// ---------------------------------------------------------------------------
__global__ void ehpb_dwconv(const float* __restrict__ x, const float* __restrict__ wgt,
                            const float* __restrict__ bias, bf16* __restrict__ outT) {
  int idx = blockIdx.x * blockDim.x + threadIdx.x;  // B*NPIX*CH, c fastest
  int c = idx & 255;
  int np = idx >> 8;
  int n = np % NPIX, b = np / NPIX;
  int hh = n / IMGW, ww = n % IMGW;
  const float* xb = x + ((long)(b * CH + c)) * NPIX;
  const float* wk = wgt + c * 9;
  float acc = bias[c];
#pragma unroll
  for (int kh = 0; kh < 3; ++kh) {
    int ih = hh + kh - 1;
    if (ih < 0 || ih >= IMGH) continue;
#pragma unroll
    for (int kw = 0; kw < 3; ++kw) {
      int iw = ww + kw - 1;
      if (iw < 0 || iw >= IMGW) continue;
      acc += wk[kh * 3 + kw] * xb[ih * IMGW + iw];
    }
  }
  outT[((long)(b * NPIX) + n) * CH + c] = (bf16)acc;
}

// xs[b,c,n] = pwout[b,c,n] + pw_b[c] + x_attn[b,n,c]
__global__ void ehpb_xs(const float* __restrict__ pwo, const float* __restrict__ pwb,
                        const float* __restrict__ xattn, float* __restrict__ xs) {
  int idx = blockIdx.x * blockDim.x + threadIdx.x;
  int n = idx % NPIX, bc = idx / NPIX;
  int c = bc % CH, b = bc / CH;
  xs[idx] = pwo[idx] + pwb[c] + xattn[((long)(b * NPIX + n)) * CH + c];
}

// sgh[b,g,n] = silu(sum_c w[g,c]*xs[b,c,n] + b[g]);  block per (n-chunk, g, b)
__global__ __launch_bounds__(256)
void ehpb_sg1(const float* __restrict__ xs, const float* __restrict__ wgt,
              const float* __restrict__ bias, float* __restrict__ sgh) {
  __shared__ float wrow[CH];
  int t = threadIdx.x, g = blockIdx.y, b = blockIdx.z;
  wrow[t] = wgt[g * CH + t];
  __syncthreads();
  int n = blockIdx.x * 256 + t;
  float acc = bias[g];
  for (int c = 0; c < CH; ++c) acc += wrow[c] * xs[((long)(b * CH + c)) * NPIX + n];
  sgh[((long)(b * GCH + g)) * NPIX + n] = siluf(acc);
}

// sg[b,n] = sigmoid(7x7 conv (32ch->1) of sgh, pad 3)
__global__ __launch_bounds__(256)
void ehpb_sg2(const float* __restrict__ sgh, const float* __restrict__ wgt,
              const float* __restrict__ bias, float* __restrict__ sg) {
  __shared__ float wt[GCH * 49];
  int t = threadIdx.x;
  for (int i = t; i < GCH * 49; i += 256) wt[i] = wgt[i];
  __syncthreads();
  int idx = blockIdx.x * 256 + t;
  int b = idx / NPIX, p = idx % NPIX;
  int hh = p / IMGW, ww = p % IMGW;
  float acc = bias[0];
  for (int g = 0; g < GCH; ++g) {
    const float* src = sgh + ((long)(b * GCH + g)) * NPIX;
#pragma unroll
    for (int kh = 0; kh < 7; ++kh) {
      int ih = hh + kh - 3;
      if (ih < 0 || ih >= IMGH) continue;
#pragma unroll
      for (int kw = 0; kw < 7; ++kw) {
        int iw = ww + kw - 3;
        if (iw < 0 || iw >= IMGW) continue;
        acc += wt[g * 49 + kh * 7 + kw] * src[ih * IMGW + iw];
      }
    }
  }
  sg[idx] = sigmoidf(acc);
}

// channel gate: mean-pool -> 1x1 (C->g) silu -> 1x1 (g->C) sigmoid. block per batch.
__global__ __launch_bounds__(256)
void ehpb_cg(const float* __restrict__ xs, const float* __restrict__ w1,
             const float* __restrict__ b1, const float* __restrict__ w2,
             const float* __restrict__ b2, float* __restrict__ cgv) {
  __shared__ float meanv[CH];
  __shared__ float hidv[GCH];
  int t = threadIdx.x, b = blockIdx.x;
  const float* row = xs + ((long)(b * CH + t)) * NPIX;
  float acc = 0.0f;
  for (int n = 0; n < NPIX; ++n) acc += row[n];
  meanv[t] = acc * (1.0f / NPIX);
  __syncthreads();
  if (t < GCH) {
    float h = b1[t];
    for (int c = 0; c < CH; ++c) h += w1[t * CH + c] * meanv[c];
    hidv[t] = siluf(h);
  }
  __syncthreads();
  float o = b2[t];
  for (int g = 0; g < GCH; ++g) o += w2[t * GCH + g] * hidv[g];
  cgv[b * CH + t] = sigmoidf(o);
}

// x2 = x + gamma1 * (xs * sg * cg)
__global__ void ehpb_combine(const float* __restrict__ x, const float* __restrict__ xs,
                             const float* __restrict__ sg, const float* __restrict__ cgv,
                             const float* __restrict__ gamma1, float* __restrict__ x2) {
  int idx = blockIdx.x * blockDim.x + threadIdx.x;
  int n = idx % NPIX, bc = idx / NPIX;
  int c = bc % CH, b = bc / CH;
  x2[idx] = x[idx] + gamma1[0] * (xs[idx] * sg[b * NPIX + n] * cgv[b * CH + c]);
}

// out = x2 + gamma2 * ff  (ff in [B,N,C] row-major, out NCHW)
__global__ void ehpb_final(const float* __restrict__ x2, const float* __restrict__ ff,
                           const float* __restrict__ gamma2, float* __restrict__ out) {
  int idx = blockIdx.x * blockDim.x + threadIdx.x;
  int n = idx % NPIX, bc = idx / NPIX;
  int c = bc % CH, b = bc / CH;
  out[idx] = x2[idx] + gamma2[0] * ff[((long)(b * NPIX + n)) * CH + c];
}

// ---------------------------------------------------------------------------
extern "C" void kernel_launch(void* const* d_in, const int* in_sizes, int n_in,
                              void* d_out, int out_size, void* d_ws, size_t ws_size,
                              hipStream_t stream) {
  const float* x      = (const float*)d_in[0];
  const float* qkv_w  = (const float*)d_in[1];
  const float* out_w  = (const float*)d_in[2];
  const float* ln1_g  = (const float*)d_in[3];
  const float* ln1_b  = (const float*)d_in[4];
  const float* ln2_g  = (const float*)d_in[5];
  const float* ln2_b  = (const float*)d_in[6];
  const float* dw_w   = (const float*)d_in[7];
  const float* dw_b   = (const float*)d_in[8];
  const float* pw_w   = (const float*)d_in[9];
  const float* pw_b   = (const float*)d_in[10];
  const float* sg1_w  = (const float*)d_in[11];
  const float* sg1_b  = (const float*)d_in[12];
  const float* sg2_w  = (const float*)d_in[13];
  const float* sg2_b  = (const float*)d_in[14];
  const float* cg1_w  = (const float*)d_in[15];
  const float* cg1_b  = (const float*)d_in[16];
  const float* cg2_w  = (const float*)d_in[17];
  const float* cg2_b  = (const float*)d_in[18];
  const float* ff_w1  = (const float*)d_in[19];
  const float* ff_b1  = (const float*)d_in[20];
  const float* ff_w2  = (const float*)d_in[21];
  const float* ff_b2  = (const float*)d_in[22];
  const float* gamma1 = (const float*)d_in[23];
  const float* gamma2 = (const float*)d_in[24];

  char* base = (char*)d_ws;
  size_t off = 0;
  auto carve = [&](size_t nbytes) {
    void* r = base + off;
    off += (nbytes + 255) & ~(size_t)255;
    return r;
  };
  bf16* wqkvb = (bf16*)carve((size_t)CH * 3 * CH * 2);    // [3C][C]  (B^T)
  bf16* woutb = (bf16*)carve((size_t)CH * CH * 2);        // [C][C]   (B^T)
  bf16* wff1b = (bf16*)carve((size_t)CH * HID * 2);       // [HID][C] (B^T)
  bf16* wff2b = (bf16*)carve((size_t)HID * CH * 2);       // [C][HID] (B^T)
  bf16* wpwb  = (bf16*)carve((size_t)CH * CH * 2);        // [co][ci] (A)
  bf16* ln1o  = (bf16*)carve((size_t)BNROWS * CH * 2);
  float* qkvf = (float*)carve((size_t)BNROWS * 3 * CH * 4);
  bf16* qb    = (bf16*)carve((size_t)NBH * NPIX * DHEAD * 2);
  bf16* kb    = (bf16*)carve((size_t)NBH * NPIX * DHEAD * 2);
  bf16* vbT   = (bf16*)carve((size_t)NBH * DHEAD * NPIX * 2);
  float* rmask = (float*)carve((size_t)NBH * NPIX * 4);
  float* cmask = (float*)carve((size_t)NBH * NPIX * 4);
  bf16* ob    = (bf16*)carve((size_t)BNROWS * CH * 2);
  float* xattn = (float*)carve((size_t)BNROWS * CH * 4);
  bf16* xcbT  = (bf16*)carve((size_t)BATCH * NPIX * CH * 2);  // [b][n][ci] (B^T)
  float* pwo  = (float*)carve((size_t)BATCH * CH * NPIX * 4);
  float* xs   = (float*)carve((size_t)BATCH * CH * NPIX * 4);
  float* sgh  = (float*)carve((size_t)BATCH * GCH * NPIX * 4);
  float* sg   = (float*)carve((size_t)BATCH * NPIX * 4);
  float* cgv  = (float*)carve((size_t)BATCH * CH * 4);
  float* x2   = (float*)carve((size_t)BATCH * CH * NPIX * 4);
  bf16* ln2o  = (bf16*)carve((size_t)BNROWS * CH * 2);
  bf16* hb    = (bf16*)carve((size_t)BNROWS * HID * 2);
  float* ff   = (float*)carve((size_t)BNROWS * CH * 4);

  // weight conversions (B operands transposed to [N][K])
  ehpb_f2bt<<<(CH * 3 * CH + 255) / 256, 256, 0, stream>>>(qkv_w, wqkvb, CH, 3 * CH);
  ehpb_f2bt<<<(CH * CH + 255) / 256, 256, 0, stream>>>(out_w, woutb, CH, CH);
  ehpb_f2bt<<<(CH * HID + 255) / 256, 256, 0, stream>>>(ff_w1, wff1b, CH, HID);
  ehpb_f2bt<<<(HID * CH + 255) / 256, 256, 0, stream>>>(ff_w2, wff2b, HID, CH);
  ehpb_f2b<<<(CH * CH + 255) / 256, 256, 0, stream>>>(pw_w, wpwb, CH * CH);

  // attention branch
  ehpb_ln_nchw<<<BNROWS, 256, 0, stream>>>(x, ln1_g, ln1_b, ln1o);
  ehpb_gemm<<<dim3(3 * CH / 128, BNROWS / 64, 1), 256, 0, stream>>>(
      ln1o, wqkvb, nullptr, qkvf, nullptr, BNROWS, 3 * CH, CH, 0, 0, 0, 0);
  ehpb_repack<<<(BNROWS * CH) / 256, 256, 0, stream>>>(qkvf, qb, kb, vbT);
  ehpb_masks<<<NBH, 256, 0, stream>>>(qb, kb, rmask, cmask);
  ehpb_attn<<<(NBH * (NPIX / 16)) / 8, 256, 0, stream>>>(qb, kb, vbT, rmask, cmask, ob);
  ehpb_gemm<<<dim3(CH / 128, BNROWS / 64, 1), 256, 0, stream>>>(
      ob, woutb, nullptr, xattn, nullptr, BNROWS, CH, CH, 0, 0, 0, 0);

  // conv branch (pointwise as batched WMMA GEMM: W[256,256] x xcT[b][2304,256]^T)
  ehpb_dwconv<<<(BATCH * NPIX * CH) / 256, 256, 0, stream>>>(x, dw_w, dw_b, xcbT);
  ehpb_gemm<<<dim3(NPIX / 128, CH / 64, BATCH), 256, 0, stream>>>(
      wpwb, xcbT, nullptr, pwo, nullptr, CH, NPIX, CH, 0,
      0, (long)NPIX * CH, (long)CH * NPIX);
  ehpb_xs<<<(BATCH * CH * NPIX) / 256, 256, 0, stream>>>(pwo, pw_b, xattn, xs);

  // gates
  ehpb_sg1<<<dim3(NPIX / 256, GCH, BATCH), 256, 0, stream>>>(xs, sg1_w, sg1_b, sgh);
  ehpb_sg2<<<(BATCH * NPIX) / 256, 256, 0, stream>>>(sgh, sg2_w, sg2_b, sg);
  ehpb_cg<<<BATCH, 256, 0, stream>>>(xs, cg1_w, cg1_b, cg2_w, cg2_b, cgv);
  ehpb_combine<<<(BATCH * CH * NPIX) / 256, 256, 0, stream>>>(x, xs, sg, cgv, gamma1, x2);

  // FFN
  ehpb_ln_nchw<<<BNROWS, 256, 0, stream>>>(x2, ln2_g, ln2_b, ln2o);
  ehpb_gemm<<<dim3(HID / 128, BNROWS / 64, 1), 256, 0, stream>>>(
      ln2o, wff1b, ff_b1, nullptr, hb, BNROWS, HID, CH, 1, 0, 0, 0);
  ehpb_gemm<<<dim3(CH / 128, BNROWS / 64, 1), 256, 0, stream>>>(
      hb, wff2b, ff_b2, ff, nullptr, BNROWS, CH, HID, 0, 0, 0, 0);
  ehpb_final<<<(BATCH * CH * NPIX) / 256, 256, 0, stream>>>(x2, ff, gamma2, (float*)d_out);
}
